// CostVolume_6201932775918
// MI455X (gfx1250) — matrix operand order repaired and checked by reading, not compiled, not run on hardware
//
#include <hip/hip_runtime.h>
#include <hip/hip_bf16.h>
#include <stdint.h>

// ---------------------------------------------------------------------------
// CDNA5 (gfx1250) wave32 WMMA implementation of the cost-volume network.
// All GEMMs run through v_wmma_f32_16x16x32_bf16 with f32 accumulation.
// Conv kernels batch 2 M-tiles per wave to halve B-fragment traffic.
// ---------------------------------------------------------------------------

typedef __attribute__((ext_vector_type(16))) __bf16   v16bf;
typedef __attribute__((ext_vector_type(8)))  __bf16   v8bf;
typedef __attribute__((ext_vector_type(8)))  float    v8f;
typedef __attribute__((ext_vector_type(8)))  uint32_t v8u;

#define CIN   1024
#define HH_   48
#define WW_   160
#define FEAT  64
#define DD_   48
#define HW    (HH_ * WW_)          // 7680
#define M3    (DD_ * HH_ * WW_)    // 368640
#define K3A   (128 * 27)           // 3456
#define K3B   (64 * 27)            // 1728
#define EPS   1e-5f
#define MT    2                    // M-tiles per wave in conv kernels

__device__ __forceinline__ __bf16 f2bf(float f) {
  uint32_t u = __builtin_bit_cast(uint32_t, f);
  uint32_t r = u + 0x7FFFu + ((u >> 16) & 1u);   // round-to-nearest-even
  uint16_t h = (uint16_t)(r >> 16);
  return __builtin_bit_cast(__bf16, h);
}

// Pack two f32 into a bf16 pair (truncation) with one v_perm_b32.
__device__ __forceinline__ uint32_t pack2bf(float lo, float hi) {
  return __builtin_amdgcn_perm(__builtin_bit_cast(uint32_t, hi),
                               __builtin_bit_cast(uint32_t, lo),
                               0x07060302u);
}

__device__ __forceinline__ v8bf zero8() {
  v8bf z;
#pragma unroll
  for (int i = 0; i < 8; ++i) z[i] = __builtin_bit_cast(__bf16, (uint16_t)0);
  return z;
}

__device__ __forceinline__ v8f zacc() {
  v8f a;
#pragma unroll
  for (int i = 0; i < 8; ++i) a[i] = 0.0f;
  return a;
}

__device__ __forceinline__ v16bf combine(v8bf lo, v8bf hi) {
  v16bf a;
#pragma unroll
  for (int i = 0; i < 8; ++i) { a[i] = lo[i]; a[i + 8] = hi[i]; }
  return a;
}

// ---------------------------------------------------------------------------
// Prep: fold BN(+bias) into per-channel scale/bias; convert + transpose
// weights to bf16 in [n][tap][cin] order (cin innermost == contiguous K).
// ---------------------------------------------------------------------------
__global__ void prep_kernel(
    const float* __restrict__ ds_w, const float* __restrict__ ds_b,
    const float* __restrict__ g2,  const float* __restrict__ b2,
    const float* __restrict__ m2,  const float* __restrict__ v2,
    const float* __restrict__ c3a_w, const float* __restrict__ c3a_b,
    const float* __restrict__ g3a, const float* __restrict__ b3a,
    const float* __restrict__ m3a, const float* __restrict__ v3a,
    const float* __restrict__ c3b_w, const float* __restrict__ c3b_b,
    const float* __restrict__ g3b, const float* __restrict__ b3b,
    const float* __restrict__ m3b, const float* __restrict__ v3b,
    __bf16* __restrict__ wb1, __bf16* __restrict__ wb3a, __bf16* __restrict__ wb3b,
    float* __restrict__ sc1,  float* __restrict__ bi1,
    float* __restrict__ sc3a, float* __restrict__ bi3a,
    float* __restrict__ sc3b, float* __restrict__ bi3b)
{
  const int tid    = blockIdx.x * blockDim.x + threadIdx.x;
  const int stride = gridDim.x * blockDim.x;

  if (tid < FEAT) {
    float s;
    s = g2[tid] * rsqrtf(v2[tid] + EPS);
    sc1[tid]  = s;  bi1[tid]  = ds_b[tid] * s + (b2[tid] - m2[tid] * s);
    s = g3a[tid] * rsqrtf(v3a[tid] + EPS);
    sc3a[tid] = s;  bi3a[tid] = c3a_b[tid] * s + (b3a[tid] - m3a[tid] * s);
    s = g3b[tid] * rsqrtf(v3b[tid] + EPS);
    sc3b[tid] = s;  bi3b[tid] = c3b_b[tid] * s + (b3b[tid] - m3b[tid] * s);
  }

  // ds_w is already [n][k] — straight convert.
  for (int i = tid; i < FEAT * CIN; i += stride) wb1[i] = f2bf(ds_w[i]);

  // c3a_w: (n, cin, kd, kh, kw) -> [n][tap][cin]
  for (int i = tid; i < FEAT * K3A; i += stride) {
    int n = i / K3A, r = i % K3A, tap = r / 128, cin = r % 128;
    int kd = tap / 9, kh = (tap / 3) % 3, kw = tap % 3;
    wb3a[i] = f2bf(c3a_w[(((n * 128 + cin) * 3 + kd) * 3 + kh) * 3 + kw]);
  }
  // c3b_w: (n, cin, kd, kh, kw) -> [n][tap][cin]
  for (int i = tid; i < FEAT * K3B; i += stride) {
    int n = i / K3B, r = i % K3B, tap = r / 64, cin = r % 64;
    int kd = tap / 9, kh = (tap / 3) % 3, kw = tap % 3;
    wb3b[i] = f2bf(c3b_w[(((n * 64 + cin) * 3 + kd) * 3 + kh) * 3 + kw]);
  }
}

// ---------------------------------------------------------------------------
// Downsample: GEMM  M=7680 (h*w), N=64 (out ch), K=1024 (in ch)
// A[m][k] = x[k*HW + m] (f32 -> bf16 via v_perm pack), fused BN+ReLU,
// output channels-last bf16 [h][w][c].
// One wave per 16-row M tile; block = 8 waves; grid = 480/8 = 60.
// All 4 B fragments are preloaded before the WMMA burst so the load
// clause can drain while the matrix pipe works (no WAR serialization).
// ---------------------------------------------------------------------------
__global__ void downsample_gemm(const float* __restrict__ x,
                                const __bf16* __restrict__ wb,
                                const float* __restrict__ sc,
                                const float* __restrict__ bi,
                                __bf16* __restrict__ out)
{
  const int lane  = threadIdx.x & 31;
  const int wave  = threadIdx.x >> 5;
  const int row   = lane & 15;          // A row / B col / D col within tile
  const int half  = lane >> 4;          // 0: K 0..7/16..23   1: K 8..15/24..31
  const int mbase = (blockIdx.x * 8 + wave) * 16;
  const int mr    = mbase + row;

  v8f acc[4];
#pragma unroll
  for (int nt = 0; nt < 4; ++nt) acc[nt] = zacc();

#pragma unroll 1
  for (int k0 = 0; k0 < CIN; k0 += 32) {
    // Preload all 4 B fragments (independent destinations -> one big clause)
    v16bf bf[4];
#pragma unroll
    for (int nt = 0; nt < 4; ++nt) {
      const v8bf* bp = reinterpret_cast<const v8bf*>(
          wb + (nt * 16 + row) * CIN + k0 + half * 8);
      bf[nt] = combine(bp[0], bp[2]);
    }

    // A fragment: dword p holds K elements 2p,2p+1 of the lane's runs.
    v8u au;
#pragma unroll
    for (int p = 0; p < 4; ++p) {
      int k = k0 + half * 8 + 2 * p;
      au[p] = pack2bf(x[k * HW + mr], x[(k + 1) * HW + mr]);
    }
#pragma unroll
    for (int p = 0; p < 4; ++p) {
      int k = k0 + 16 + half * 8 + 2 * p;
      au[p + 4] = pack2bf(x[k * HW + mr], x[(k + 1) * HW + mr]);
    }
    v16bf a = __builtin_bit_cast(v16bf, au);

#pragma unroll
    for (int nt = 0; nt < 4; ++nt) {
      acc[nt] = __builtin_amdgcn_wmma_f32_16x16x32_bf16(
          false, a, false, bf[nt], (short)0, acc[nt], false, false);
    }
  }

#pragma unroll
  for (int nt = 0; nt < 4; ++nt) {
    const int c = nt * 16 + row;
    const float s = sc[c], t = bi[c];
#pragma unroll
    for (int e = 0; e < 8; ++e) {
      int m = mbase + half * 8 + e;
      float v = fmaxf(acc[nt][e] * s + t, 0.0f);
      out[m * FEAT + c] = f2bf(v);
    }
  }
}

// ---------------------------------------------------------------------------
// conv3a: implicit GEMM with fused cost-volume construction.
// M = D*H*W = 368640 output points, N = 64, K = 27 taps * 128 cin.
// cost[c,d,h,w] = (w>=d) ? (c<64 ? left[c,h,w] : right[c-64,h,w-d]) : 0
// Inputs channels-last bf16; zero-pad + mask via predicated A loads.
// Each wave computes MT=2 M-tiles so B fragments are reused.
// ---------------------------------------------------------------------------
__global__ void conv3a_kernel(const __bf16* __restrict__ L,
                              const __bf16* __restrict__ R,
                              const __bf16* __restrict__ wb,
                              const float* __restrict__ sc,
                              const float* __restrict__ bi,
                              __bf16* __restrict__ y1)
{
  const int lane = threadIdx.x & 31;
  const int wave = threadIdx.x >> 5;
  const int row  = lane & 15;
  const int half = lane >> 4;
  const int mt0  = (blockIdx.x * 8 + wave) * MT;   // first of MT tile indices

  int mb[MT], wv[MT], hv[MT], dv[MT];
#pragma unroll
  for (int mt = 0; mt < MT; ++mt) {
    mb[mt] = (mt0 + mt) * 16;
    wv[mt] = (mb[mt] % WW_) + row;        // tile = 16 consecutive w
    hv[mt] = (mb[mt] / WW_) % HH_;
    dv[mt] = mb[mt] / (WW_ * HH_);
  }

  v8f acc[MT][4];
#pragma unroll
  for (int mt = 0; mt < MT; ++mt)
#pragma unroll
    for (int nt = 0; nt < 4; ++nt) acc[mt][nt] = zacc();

#pragma unroll 1
  for (int tap = 0; tap < 27; ++tap) {
    const int kd = tap / 9, kh = (tap / 3) % 3, kw = tap % 3;

    bool inb[MT];
    int  la[MT], ra[MT];
#pragma unroll
    for (int mt = 0; mt < MT; ++mt) {
      const int dd = dv[mt] + kd - 1, hh = hv[mt] + kh - 1, ww = wv[mt] + kw - 1;
      inb[mt] = ((unsigned)dd < (unsigned)DD_) &&
                ((unsigned)hh < (unsigned)HH_) &&
                ((unsigned)ww < (unsigned)WW_) && (ww >= dd);
      la[mt]  = (hh * WW_ + ww) * FEAT;            // left  @ (hh, ww)
      ra[mt]  = (hh * WW_ + (ww - dd)) * FEAT;     // right @ (hh, ww-dd)
    }

    if (tap < 26)  // warm caches for next tap's weights
      __builtin_prefetch(wb + row * K3A + (tap + 1) * 128, 0, 3);

#pragma unroll
    for (int c0 = 0; c0 < 128; c0 += 32) {
      // B fragments for all 4 N-tiles (shared by both M-tiles)
      v16bf bf[4];
#pragma unroll
      for (int nt = 0; nt < 4; ++nt) {
        const v8bf* bp = reinterpret_cast<const v8bf*>(
            wb + (nt * 16 + row) * K3A + tap * 128 + c0 + half * 8);
        bf[nt] = combine(bp[0], bp[2]);
      }
#pragma unroll
      for (int mt = 0; mt < MT; ++mt) {
        const int cA = c0 + half * 8;      // first 8-channel run
        const int cB = cA + 16;            // second run (same side of 64-split)
        const __bf16* pA = (cA < 64) ? (L + la[mt] + cA) : (R + ra[mt] + (cA - 64));
        const __bf16* pB = (cB < 64) ? (L + la[mt] + cB) : (R + ra[mt] + (cB - 64));
        v8bf r0 = inb[mt] ? *reinterpret_cast<const v8bf*>(pA) : zero8();
        v8bf r1 = inb[mt] ? *reinterpret_cast<const v8bf*>(pB) : zero8();
        v16bf a = combine(r0, r1);
#pragma unroll
        for (int nt = 0; nt < 4; ++nt) {
          acc[mt][nt] = __builtin_amdgcn_wmma_f32_16x16x32_bf16(
              false, a, false, bf[nt], (short)0, acc[mt][nt], false, false);
        }
      }
    }
  }

#pragma unroll
  for (int mt = 0; mt < MT; ++mt)
#pragma unroll
    for (int nt = 0; nt < 4; ++nt) {
      const int c = nt * 16 + row;
      const float s = sc[c], t = bi[c];
#pragma unroll
      for (int e = 0; e < 8; ++e) {
        int m = mb[mt] + half * 8 + e;
        float v = fmaxf(acc[mt][nt][e] * s + t, 0.0f);
        y1[m * FEAT + c] = f2bf(v);
      }
    }
}

// ---------------------------------------------------------------------------
// conv3b: implicit GEMM, K = 27 taps * 64 cin, input y1 bf16 channels-last,
// output f32 directly in the reference's reshape layout: out[c*M3 + m].
// ---------------------------------------------------------------------------
__global__ void conv3b_kernel(const __bf16* __restrict__ X,
                              const __bf16* __restrict__ wb,
                              const float* __restrict__ sc,
                              const float* __restrict__ bi,
                              float* __restrict__ out)
{
  const int lane = threadIdx.x & 31;
  const int wave = threadIdx.x >> 5;
  const int row  = lane & 15;
  const int half = lane >> 4;
  const int mt0  = (blockIdx.x * 8 + wave) * MT;

  int mb[MT], wv[MT], hv[MT], dv[MT];
#pragma unroll
  for (int mt = 0; mt < MT; ++mt) {
    mb[mt] = (mt0 + mt) * 16;
    wv[mt] = (mb[mt] % WW_) + row;
    hv[mt] = (mb[mt] / WW_) % HH_;
    dv[mt] = mb[mt] / (WW_ * HH_);
  }

  v8f acc[MT][4];
#pragma unroll
  for (int mt = 0; mt < MT; ++mt)
#pragma unroll
    for (int nt = 0; nt < 4; ++nt) acc[mt][nt] = zacc();

#pragma unroll 1
  for (int tap = 0; tap < 27; ++tap) {
    const int kd = tap / 9, kh = (tap / 3) % 3, kw = tap % 3;

    bool inb[MT];
    int  xa[MT];
#pragma unroll
    for (int mt = 0; mt < MT; ++mt) {
      const int dd = dv[mt] + kd - 1, hh = hv[mt] + kh - 1, ww = wv[mt] + kw - 1;
      inb[mt] = ((unsigned)dd < (unsigned)DD_) &&
                ((unsigned)hh < (unsigned)HH_) &&
                ((unsigned)ww < (unsigned)WW_);
      xa[mt]  = ((dd * HH_ + hh) * WW_ + ww) * FEAT;
    }

    if (tap < 26)
      __builtin_prefetch(wb + row * K3B + (tap + 1) * 64, 0, 3);

#pragma unroll
    for (int c0 = 0; c0 < 64; c0 += 32) {
      v16bf bf[4];
#pragma unroll
      for (int nt = 0; nt < 4; ++nt) {
        const v8bf* bp = reinterpret_cast<const v8bf*>(
            wb + (nt * 16 + row) * K3B + tap * 64 + c0 + half * 8);
        bf[nt] = combine(bp[0], bp[2]);
      }
#pragma unroll
      for (int mt = 0; mt < MT; ++mt) {
        const int cA = c0 + half * 8;
        const __bf16* pA = X + xa[mt] + cA;
        const __bf16* pB = X + xa[mt] + cA + 16;
        v8bf r0 = inb[mt] ? *reinterpret_cast<const v8bf*>(pA) : zero8();
        v8bf r1 = inb[mt] ? *reinterpret_cast<const v8bf*>(pB) : zero8();
        v16bf a = combine(r0, r1);
#pragma unroll
        for (int nt = 0; nt < 4; ++nt) {
          acc[mt][nt] = __builtin_amdgcn_wmma_f32_16x16x32_bf16(
              false, a, false, bf[nt], (short)0, acc[mt][nt], false, false);
        }
      }
    }
  }

#pragma unroll
  for (int mt = 0; mt < MT; ++mt)
#pragma unroll
    for (int nt = 0; nt < 4; ++nt) {
      const int c = nt * 16 + row;
      const float s = sc[c], t = bi[c];
#pragma unroll
      for (int e = 0; e < 8; ++e) {
        int m = mb[mt] + half * 8 + e;
        out[c * M3 + m] = fmaxf(acc[mt][nt][e] * s + t, 0.0f);
      }
    }
}

// ---------------------------------------------------------------------------
// Host-side orchestration
// ---------------------------------------------------------------------------
extern "C" void kernel_launch(void* const* d_in, const int* in_sizes, int n_in,
                              void* d_out, int out_size, void* d_ws, size_t ws_size,
                              hipStream_t stream) {
  const float* left   = (const float*)d_in[0];
  const float* right  = (const float*)d_in[1];
  const float* ds_w   = (const float*)d_in[2];
  const float* ds_b   = (const float*)d_in[3];
  const float* g2     = (const float*)d_in[4];
  const float* b2     = (const float*)d_in[5];
  const float* m2     = (const float*)d_in[6];
  const float* v2     = (const float*)d_in[7];
  const float* c3a_w  = (const float*)d_in[8];
  const float* c3a_b  = (const float*)d_in[9];
  const float* g3a    = (const float*)d_in[10];
  const float* b3a    = (const float*)d_in[11];
  const float* m3a    = (const float*)d_in[12];
  const float* v3a    = (const float*)d_in[13];
  const float* c3b_w  = (const float*)d_in[14];
  const float* c3b_b  = (const float*)d_in[15];
  const float* g3b    = (const float*)d_in[16];
  const float* b3b    = (const float*)d_in[17];
  const float* m3b    = (const float*)d_in[18];
  const float* v3b    = (const float*)d_in[19];

  // Workspace carve-up (all 256B aligned)
  char*  ws  = (char*)d_ws;
  size_t off = 0;
  auto carve = [&](size_t bytes) -> void* {
    void* p = ws + off;
    off += (bytes + 255) & ~(size_t)255;
    return p;
  };
  __bf16* wb1   = (__bf16*)carve((size_t)FEAT * CIN * 2);
  __bf16* wb3a  = (__bf16*)carve((size_t)FEAT * K3A * 2);
  __bf16* wb3b  = (__bf16*)carve((size_t)FEAT * K3B * 2);
  float*  sc1   = (float*)carve(FEAT * 4);
  float*  bi1   = (float*)carve(FEAT * 4);
  float*  sc3a  = (float*)carve(FEAT * 4);
  float*  bi3a  = (float*)carve(FEAT * 4);
  float*  sc3b  = (float*)carve(FEAT * 4);
  float*  bi3b  = (float*)carve(FEAT * 4);
  __bf16* Lb    = (__bf16*)carve((size_t)HW * FEAT * 2);
  __bf16* Rb    = (__bf16*)carve((size_t)HW * FEAT * 2);
  __bf16* y1    = (__bf16*)carve((size_t)M3 * FEAT * 2);
  (void)ws_size; (void)in_sizes; (void)n_in; (void)out_size;

  prep_kernel<<<512, 256, 0, stream>>>(
      ds_w, ds_b, g2, b2, m2, v2,
      c3a_w, c3a_b, g3a, b3a, m3a, v3a,
      c3b_w, c3b_b, g3b, b3b, m3b, v3b,
      wb1, wb3a, wb3b, sc1, bi1, sc3a, bi3a, sc3b, bi3b);

  // 480 M-tiles / 8 waves per block = 60 blocks
  downsample_gemm<<<60, 256, 0, stream>>>(left,  wb1, sc1, bi1, Lb);
  downsample_gemm<<<60, 256, 0, stream>>>(right, wb1, sc1, bi1, Rb);

  // 23040 M-tiles / (8 waves * MT tiles) = 1440 blocks
  conv3a_kernel<<<1440, 256, 0, stream>>>(Lb, Rb, wb3a, sc3a, bi3a, y1);
  conv3b_kernel<<<1440, 256, 0, stream>>>(y1, wb3b, sc3b, bi3b, (float*)d_out);
}
